// SparseResBlockDownsample3d_14568529068651
// MI455X (gfx1250) — compile-verified, matrix operand order checked by reading, not run on hardware
//
#include <hip/hip_runtime.h>
#include <hip/hip_bf16.h>

typedef __attribute__((ext_vector_type(16))) __bf16 v16bf;
typedef __attribute__((ext_vector_type(8)))  __bf16 v8bf;
typedef __attribute__((ext_vector_type(8)))  float  v8f;
typedef __attribute__((ext_vector_type(4)))  float  v4f;

#define DEV __device__ __forceinline__

#if defined(__has_builtin)
#if __has_builtin(__builtin_amdgcn_sched_group_barrier)
#define SCHED_GROUP(mask, size) __builtin_amdgcn_sched_group_barrier((mask), (size), 0)
#endif
#endif
#ifndef SCHED_GROUP
#define SCHED_GROUP(mask, size)
#endif

static DEV float silu_f(float x) { return x / (1.0f + __expf(-x)); }

// ---------------------------------------------------------------------------
// Stage 1: LayerNorm(affine) + SiLU + mask, then 2x2x2 masked-mean downsample.
// One wave (32 lanes) per input voxel; 8 waves per block = one output cell.
// ---------------------------------------------------------------------------
__global__ void k_stage1(const float* __restrict__ feats,
                         const unsigned char* __restrict__ mask,
                         const float* __restrict__ gamma,
                         const float* __restrict__ beta,
                         __bf16* __restrict__ h_ds,
                         __bf16* __restrict__ x_ds,
                         float* __restrict__ mask2)
{
    __shared__ float sh[8][64];
    __shared__ float sx[8][64];
    __shared__ float scnt[8];

    const int cell = blockIdx.x;                 // b*32768 + z*1024 + y*32 + x
    const int cx = cell & 31;
    const int cy = (cell >> 5) & 31;
    const int cz = (cell >> 10) & 31;
    const int cb = cell >> 15;

    const int w    = threadIdx.x >> 5;           // wave id 0..7 -> (dz,dy,dx)
    const int lane = threadIdx.x & 31;
    const int dx = w & 1, dy = (w >> 1) & 1, dz = (w >> 2) & 1;
    const int ix = cx * 2 + dx, iy = cy * 2 + dy, iz = cz * 2 + dz;

    const long vox = ((((long)cb * 64 + iz) * 64 + iy) * 64 + ix);
    const float* f = feats + vox * 64;
    const float m = mask[vox] ? 1.0f : 0.0f;

    const float2 v2 = *(const float2*)(f + lane * 2);
    float x0 = v2.x, x1 = v2.y;
    float s  = x0 + x1;
    float sq = x0 * x0 + x1 * x1;
    #pragma unroll
    for (int off = 16; off; off >>= 1) {
        s  += __shfl_xor(s,  off, 32);
        sq += __shfl_xor(sq, off, 32);
    }
    const float mu  = s * (1.0f / 64.0f);
    const float var = sq * (1.0f / 64.0f) - mu * mu;
    const float r   = rsqrtf(var + 1e-6f);

    const float h0 = silu_f((x0 - mu) * r * gamma[lane * 2 + 0] + beta[lane * 2 + 0]) * m;
    const float h1 = silu_f((x1 - mu) * r * gamma[lane * 2 + 1] + beta[lane * 2 + 1]) * m;

    sh[w][lane * 2 + 0] = h0;
    sh[w][lane * 2 + 1] = h1;
    sx[w][lane * 2 + 0] = x0 * m;
    sx[w][lane * 2 + 1] = x1 * m;
    if (lane == 0) scnt[w] = m;
    __syncthreads();

    const int c = threadIdx.x;
    if (c < 64) {
        float hs = 0.f, xs = 0.f, cnt = 0.f;
        #pragma unroll
        for (int i = 0; i < 8; ++i) { hs += sh[i][c]; xs += sx[i][c]; cnt += scnt[i]; }
        const float inv = 1.0f / fmaxf(cnt, 1.0f);
        h_ds[(long)cell * 64 + c] = (__bf16)(hs * inv);
        x_ds[(long)cell * 64 + c] = (__bf16)(xs * inv);
        if (c == 0) mask2[cell] = (cnt > 0.f) ? 1.0f : 0.0f;
    }
}

// ---------------------------------------------------------------------------
// Weight repack: w[ntaps][cin][128] f32 -> bf16 WMMA B-fragments laid out as
// [tap][kc][nt][lane(32)][half(16)] per the 16-bit B VGPR striping.
// ---------------------------------------------------------------------------
__global__ void k_pack(const float* __restrict__ w, __bf16* __restrict__ wp,
                       int ntaps, int cin)
{
    const int idx = blockIdx.x * 256 + threadIdx.x;
    const int kcN = cin / 32;
    const int total = ntaps * kcN * 8 * 32 * 16;
    if (idx >= total) return;
    const int h    = idx & 15;
    const int lane = (idx >> 4) & 31;
    const int nt   = (idx >> 9) & 7;
    const int rest = idx >> 12;              // tap*kcN + kc
    const int kc   = rest % kcN;
    const int tap  = rest / kcN;
    const int k = kc * 32 + (h < 8 ? h : h + 8) + (lane >= 16 ? 8 : 0);
    const int n = nt * 16 + (lane & 15);
    wp[idx] = (__bf16)(w[((long)tap * cin + k) * 128 + n]);
}

// ---------------------------------------------------------------------------
// Async global->LDS staging of one tap's packed weights (bytes = KC*8192).
// 256 threads, each thread issues KC*2 async b128 copies. ASYNCcnt-tracked.
// ---------------------------------------------------------------------------
template<int KC>
static DEV void stage_tap_async(const __bf16* wp_tap, unsigned lds_base, int tid)
{
    #pragma unroll
    for (int j = 0; j < KC * 2; ++j) {
        const int chunk = tid + j * 256;                 // 16-byte chunks
        const unsigned      ldsa = lds_base + (unsigned)chunk * 16u;
        const void*         gsrc = (const char*)wp_tap + (size_t)chunk * 16u;
        asm volatile("global_load_async_to_lds_b128 %0, %1, off"
                     :: "v"(ldsa), "v"(gsrc) : "memory");
    }
}

static DEV void wait_async_done()
{
    asm volatile("s_wait_asynccnt 0x0" ::: "memory");
}

// ---------------------------------------------------------------------------
// Implicit-GEMM conv via WMMA bf16 (f32 accumulate).
// 8 waves/block; one wave per tile of 16 voxels (contiguous in W) x 128 chans.
// Per-tap weights staged through LDS with double-buffered async copies; B
// fragments consumed through group-level double buffering, with
// sched_group_barrier pinning DS-reads of group g+1 ahead of group g's WMMAs.
// ---------------------------------------------------------------------------
template<int CIN, bool ADD_SKIP, bool IS27>
__global__ void k_conv(const __bf16* __restrict__ act,   // [131072][CIN]
                       const __bf16* __restrict__ wp,    // packed fragments
                       const float* __restrict__ bias,   // [128]
                       const float* __restrict__ mask2,  // [131072]
                       const float* __restrict__ skip,   // [131072][128] or null
                       float* __restrict__ out)          // [131072][128]
{
    constexpr int KC = CIN / 32;
    constexpr int TAP_HALVES = KC * 8 * 512;             // bf16 elems per tap tile
    __shared__ __bf16 blds[2][TAP_HALVES];

    const int tid    = threadIdx.x;
    const int wave   = blockIdx.x * 8 + (tid >> 5);
    const int lane   = tid & 31;
    const int laneM  = lane & 15;
    const int laneHi = lane >> 4;            // 0: K low halves, 1: K high halves

    // tile decode: wave = b*2048 + z*64 + y*2 + xhalf
    const int x0 = (wave & 1) * 16;
    const int y  = (wave >> 1) & 31;
    const int z  = (wave >> 6) & 31;
    const int b  = wave >> 11;
    const int x  = x0 + laneM;
    const long voxBase = (((long)b * 32 + z) * 32 + y) * 32;

    v8f acc[8];
    #pragma unroll
    for (int i = 0; i < 8; ++i)
        #pragma unroll
        for (int j = 0; j < 8; ++j) acc[i][j] = 0.0f;

    const unsigned lds0 = (unsigned)(uintptr_t)&blds[0][0];
    const unsigned lds1 = (unsigned)(uintptr_t)&blds[1][0];

    const int ntaps = IS27 ? 27 : 1;

    // prime the pipeline: tap 0 -> buffer 0
    stage_tap_async<KC>(wp, lds0, tid);

    for (int t = 0; t < ntaps; ++t) {
        wait_async_done();           // this wave's copies into buf[t&1] landed
        __syncthreads();             // all waves' copies landed; prev compute done

        if (t + 1 < ntaps)
            stage_tap_async<KC>(wp + (long)(t + 1) * TAP_HALVES,
                                ((t + 1) & 1) ? lds1 : lds0, tid);

        const int dz = IS27 ? (t / 9) - 1 : 0;
        const int dy = IS27 ? ((t / 3) % 3) - 1 : 0;
        const int dx = IS27 ? (t % 3) - 1 : 0;
        const int zz = z + dz, yy = y + dy, xx = x + dx;
        const bool ok = ((unsigned)zz < 32u) && ((unsigned)yy < 32u) && ((unsigned)xx < 32u);
        const long nvox = (((long)b * 32 + zz) * 32 + yy) * 32 + xx;
        const __bf16* ap = act + nvox * CIN + laneHi * 8;

        // warm next tap's activation row (gfx1250 global_prefetch_b8)
        if (IS27 && t + 1 < 27)
            __builtin_prefetch((const void*)(ap + (long)CIN), 0, 3);

        v16bf afrag[KC];
        #pragma unroll
        for (int kc = 0; kc < KC; ++kc) {
            v8bf alo, ahi;
            #pragma unroll
            for (int j = 0; j < 8; ++j) { alo[j] = (__bf16)0.0f; ahi[j] = (__bf16)0.0f; }
            if (ok) {
                alo = *(const v8bf*)(ap + kc * 32);
                ahi = *(const v8bf*)(ap + kc * 32 + 16);
            }
            v16bf a;
            #pragma unroll
            for (int j = 0; j < 8; ++j) { a[j] = alo[j]; a[j + 8] = ahi[j]; }
            afrag[kc] = a;
        }

        // B fragments: group-level double buffering over the 8 n-tiles.
        // group g holds KC fragments; load group g+1 while WMMA-ing group g.
        const __bf16* bb = &blds[t & 1][0];
        v16bf bg[2][KC];
        #pragma unroll
        for (int kc = 0; kc < KC; ++kc)
            bg[0][kc] = *(const v16bf*)(bb + (kc * 8 + 0) * 512 + lane * 16);

        #pragma unroll
        for (int g = 0; g < 8; ++g) {
            const int cur = g & 1;
            const int nxt = cur ^ 1;
            if (g + 1 < 8) {
                #pragma unroll
                for (int kc = 0; kc < KC; ++kc)
                    bg[nxt][kc] = *(const v16bf*)(bb + (kc * 8 + (g + 1)) * 512
                                                  + lane * 16);
            }
            #pragma unroll
            for (int kc = 0; kc < KC; ++kc) {
                acc[g] = __builtin_amdgcn_wmma_f32_16x16x32_bf16(
                    false, afrag[kc], false, bg[cur][kc], (short)0, acc[g],
                    false, false);
            }
            // pin schedule: next group's DS reads ahead of this group's WMMAs
            SCHED_GROUP(0x100, 2 * KC);   // DS read
            SCHED_GROUP(0x008, KC);       // MFMA/WMMA
        }
    }

    // D layout: vgpr r -> lanes 0-15: M=r, N=lane; lanes 16-31: M=r+8, N=lane-16
    #pragma unroll
    for (int nt = 0; nt < 8; ++nt) {
        const int n  = nt * 16 + laneM;
        const float bv = bias[n];
        #pragma unroll
        for (int r = 0; r < 8; ++r) {
            const int  mloc = r + laneHi * 8;
            const long vox  = voxBase + x0 + mloc;
            const float mv  = mask2[vox];
            float v = (acc[nt][r] + bv) * mv;
            if (ADD_SKIP) v += skip[vox * 128 + n];
            out[vox * 128 + n] = v;
        }
    }
}

// ---------------------------------------------------------------------------
// norm2: LayerNorm (no affine) over 128 channels + SiLU + mask -> bf16.
// One wave per voxel, 4 channels per lane.
// ---------------------------------------------------------------------------
__global__ void k_norm2(const float* __restrict__ h1,
                        const float* __restrict__ mask2,
                        __bf16* __restrict__ h2)
{
    const int vox  = blockIdx.x * (blockDim.x >> 5) + (threadIdx.x >> 5);
    const int lane = threadIdx.x & 31;
    const float* p = h1 + (long)vox * 128 + lane * 4;
    const v4f v = *(const v4f*)p;
    float s  = v[0] + v[1] + v[2] + v[3];
    float sq = v[0] * v[0] + v[1] * v[1] + v[2] * v[2] + v[3] * v[3];
    #pragma unroll
    for (int off = 16; off; off >>= 1) {
        s  += __shfl_xor(s,  off, 32);
        sq += __shfl_xor(sq, off, 32);
    }
    const float mu  = s * (1.0f / 128.0f);
    const float var = sq * (1.0f / 128.0f) - mu * mu;
    const float r   = rsqrtf(var + 1e-6f);
    const float m   = mask2[vox];
    __bf16* q = h2 + (long)vox * 128 + lane * 4;
    #pragma unroll
    for (int i = 0; i < 4; ++i)
        q[i] = (__bf16)(silu_f((v[i] - mu) * r) * m);
}

// ---------------------------------------------------------------------------
extern "C" void kernel_launch(void* const* d_in, const int* in_sizes, int n_in,
                              void* d_out, int out_size, void* d_ws, size_t ws_size,
                              hipStream_t stream)
{
    const float*         feats  = (const float*)d_in[0];
    const unsigned char* maskb  = (const unsigned char*)d_in[1];  // jax bool = 1 byte
    const float*         gamma1 = (const float*)d_in[2];
    const float*         beta1  = (const float*)d_in[3];
    const float*         w1     = (const float*)d_in[4];
    const float*         b1     = (const float*)d_in[5];
    const float*         w2     = (const float*)d_in[6];
    const float*         b2     = (const float*)d_in[7];
    const float*         wsk    = (const float*)d_in[8];
    const float*         bsk    = (const float*)d_in[9];
    float*               outp   = (float*)d_out;

    const long NVOX = 4L * 32 * 32 * 32;   // 131072 downsampled voxels

    char* ws = (char*)d_ws;
    size_t off = 0;
    auto alloc = [&](size_t bytes) -> char* {
        char* p = ws + off;
        off += (bytes + 255) & ~(size_t)255;
        return p;
    };
    __bf16* h_ds  = (__bf16*)alloc(NVOX * 64 * 2);
    __bf16* x_ds  = (__bf16*)alloc(NVOX * 64 * 2);
    float*  mask2 = (float*) alloc(NVOX * 4);
    float*  h1    = (float*) alloc(NVOX * 128 * 4);
    float*  skipb = (float*) alloc(NVOX * 128 * 4);
    __bf16* h2    = (__bf16*)alloc(NVOX * 128 * 2);
    __bf16* w1p   = (__bf16*)alloc((size_t)27 * 2 * 8 * 512 * 2);
    __bf16* w2p   = (__bf16*)alloc((size_t)27 * 4 * 8 * 512 * 2);
    __bf16* wskp  = (__bf16*)alloc((size_t)1 * 2 * 8 * 512 * 2);

    // 1) LN1 + SiLU + mask + downsample (also x_ds, mask2)
    k_stage1<<<131072, 256, 0, stream>>>(feats, maskb, gamma1, beta1, h_ds, x_ds, mask2);

    // 2) pack weights into WMMA B-fragment layout
    k_pack<<<(27 * 2 * 8 * 512 + 255) / 256, 256, 0, stream>>>(w1,  w1p, 27, 64);
    k_pack<<<(27 * 4 * 8 * 512 + 255) / 256, 256, 0, stream>>>(w2,  w2p, 27, 128);
    k_pack<<<(1  * 2 * 8 * 512 + 255) / 256, 256, 0, stream>>>(wsk, wskp, 1, 64);

    // 3) conv1 (64 -> 128), masked, f32 out
    k_conv<64, false, true><<<1024, 256, 0, stream>>>(h_ds, w1p, b1, mask2, nullptr, h1);

    // 4) skip GEMM: x_ds @ w_skip + b_skip, masked
    k_conv<64, false, false><<<1024, 256, 0, stream>>>(x_ds, wskp, bsk, mask2, nullptr, skipb);

    // 5) norm2 + SiLU + mask -> bf16
    k_norm2<<<16384, 256, 0, stream>>>(h1, mask2, h2);

    // 6) conv2 (128 -> 128) + bias, masked, + skip -> out
    k_conv<128, true, true><<<1024, 256, 0, stream>>>(h2, w2p, b2, mask2, skipb, outp);

    (void)in_sizes; (void)n_in; (void)out_size; (void)ws_size;
}
